// BitwiseResnet18_18648747999436
// MI455X (gfx1250) — 8-bit matrix operand order repaired and checked by reading, compiled, not run on hardware
//
#include <hip/hip_runtime.h>
#include <hip/hip_bf16.h>


typedef int v8i __attribute__((ext_vector_type(8)));

#define BRN_EPS 1e-5f

__global__ void brn_fold_bn(const float* __restrict__ g, const float* __restrict__ b,
                            const float* __restrict__ m, const float* __restrict__ v,
                            int C, float* __restrict__ sc, float* __restrict__ sh) {
    int c = blockIdx.x * 64 + threadIdx.x;
    if (c >= C) return;
    float inv = rsqrtf(v[c] + BRN_EPS);
    float s = g[c] * inv;
    sc[c] = s;
    sh[c] = b[c] - m[c] * s;
}

__global__ void brn_pack_w(const float* __restrict__ w, signed char* __restrict__ out,
                           int Cout, int Cin, int ks2, int coShift, int nchShift) {
    long id = (long)blockIdx.x * 256 + threadIdx.x;
    long total = (long)Cout * Cin * ks2;
    if (id >= total) return;
    int kl  = (int)(id & 63);
    int co  = ((int)(id >> 6)) & (Cout - 1);
    int rest = (int)(id >> (6 + coShift));
    int ch  = rest & ((1 << nchShift) - 1);
    int tap = rest >> nchShift;
    int ci  = (ch << 6) + kl;
    float val = w[((long)co * Cin + ci) * ks2 + tap];
    out[id] = (val >= 0.0f) ? (signed char)1 : (signed char)-1;
}

__global__ void brn_stem(const float* __restrict__ x, const float* __restrict__ w,
                         const float* __restrict__ sc, const float* __restrict__ sh,
                         float* __restrict__ outf, signed char* __restrict__ outs) {
    long gid = (long)blockIdx.x * 256 + threadIdx.x;
    int xo = (int)(gid & 31);
    int y  = (int)((gid >> 5) & 31);
    int c  = (int)((gid >> 10) & 63);
    long n = gid >> 16;
    float s = 0.0f;
    for (int ci = 0; ci < 3; ++ci)
        for (int ky = 0; ky < 7; ++ky) {
            int yi = y + ky - 3;
            if (yi < 0 || yi > 31) continue;
            for (int kx = 0; kx < 7; ++kx) {
                int xi = xo + kx - 3;
                if (xi < 0 || xi > 31) continue;
                s += x[((n * 3 + ci) * 32 + yi) * 32 + xi] *
                     w[((c * 3 + ci) * 7 + ky) * 7 + kx];
            }
        }
    float f = s * sc[c] + sh[c];
    outf[gid] = f;
    outs[gid] = (f >= 0.0f) ? (signed char)1 : (signed char)-1;
}

template <int NCH, int KS>
__global__ __launch_bounds__(128) void brn_binconv_wmma(
    const signed char* __restrict__ act,
    const signed char* __restrict__ wpack,
    const float* __restrict__ bnsc, const float* __restrict__ bnsh,
    const float* __restrict__ idn,
    float* __restrict__ outf,
    signed char* __restrict__ outs,
    int Cin, int Hin, int Win, int Cout,
    int hwoShift, int woShift, int stride, int pad)
{
    __shared__ int4 Bt4[4][64];
    char* Bt = (char*)Bt4;

    const int tid  = threadIdx.x;
    const int lane = tid & 31;
    const int mt   = blockIdx.y * 4 + (tid >> 5);
    const int co0  = mt * 16;
    const int ptile = blockIdx.x;
    const int HWin = Hin * Win;
    const int HWo  = 1 << hwoShift;
    const int Wout = 1 << woShift;
    constexpr int TAPS = KS * KS;

    const int fcol = tid >> 3;
    const int fk0  = (tid & 7) * 8;
    int nf[2], yf[2], xf[2];
    #pragma unroll
    for (int nt = 0; nt < 2; ++nt) {
        int p  = ptile * 32 + nt * 16 + fcol;
        nf[nt] = p >> hwoShift;
        int rm = p & (HWo - 1);
        yf[nt] = rm >> woShift;
        xf[nt] = rm & (Wout - 1);
    }

    const unsigned ldsBase =
        (unsigned)(unsigned long long)(__attribute__((address_space(3))) char*)Bt;

    auto issue_fill = [&](int tapn, int chn, int buf) {
        const int ky = tapn / KS, kx = tapn % KS;
        #pragma unroll
        for (int nt = 0; nt < 2; ++nt) {
            const int yi = yf[nt] * stride + ky - pad;
            const int xi = xf[nt] * stride + kx - pad;
            const bool inb = ((unsigned)yi < (unsigned)Hin) &&
                             ((unsigned)xi < (unsigned)Win);
            const int seg = (buf * 2 + nt) * 1024 + tid * 8;
            if (inb) {
                const signed char* gp0 = act +
                    (nf[nt] * Cin + (chn * 64 + fk0)) * HWin + yi * Win + xi;
                const unsigned dst = ldsBase + (unsigned)seg;
                #pragma unroll
                for (int j = 0; j < 8; ++j) {
                    asm volatile("global_load_async_to_lds_b8 %0, %1, off"
                                 :: "v"(dst + (unsigned)j),
                                    "v"((unsigned long long)(gp0 + j * HWin))
                                 : "memory");
                }
            } else {
                *(int2*)(Bt + seg) = make_int2(0, 0);
            }
        }
    };

    v8i acc[2] = {};
    int buf = 0;
    issue_fill(0, 0, 0);

    for (int tap = 0; tap < TAPS; ++tap) {
        #pragma unroll
        for (int ch = 0; ch < NCH; ++ch) {
            asm volatile("s_wait_asynccnt 0x0" ::: "memory");
            __syncthreads();

            const int tapN = (ch == NCH - 1) ? (tap + 1) : tap;
            const int chN  = (ch == NCH - 1) ? 0 : (ch + 1);
            issue_fill(tapN, chN, buf ^ 1);

            const signed char* arow = wpack +
                (((tap * NCH + ch) * Cout) + co0 + (lane & 15)) * 64 +
                ((lane & 16) ? 16 : 0);
            if (tapN < TAPS)
                __builtin_prefetch(wpack + (((tapN * NCH + chN) * Cout) + co0) * 64,
                                   0, 0);
            int4 a0 = *(const int4*)(arow);
            int4 a1 = *(const int4*)(arow + 32);
            v8i A = { a0.x, a0.y, a0.z, a0.w, a1.x, a1.y, a1.z, a1.w };

            #pragma unroll
            for (int nt = 0; nt < 2; ++nt) {
                const char* bb = Bt + (buf * 2 + nt) * 1024 +
                                 (lane & 15) * 64 + ((lane & 16) ? 16 : 0);
                int4 b0 = *(const int4*)(bb);
                int4 b1 = *(const int4*)(bb + 32);
                v8i B = { b0.x, b0.y, b0.z, b0.w, b1.x, b1.y, b1.z, b1.w };
                acc[nt] = __builtin_amdgcn_wmma_i32_16x16x64_iu8(
                    true, A, true, B, acc[nt], false, false);
            }
            buf ^= 1;
        }
    }

    const int col  = lane & 15;
    const int half = lane >> 4;
    #pragma unroll
    for (int nt = 0; nt < 2; ++nt) {
        const int p  = ptile * 32 + nt * 16 + col;
        const int n  = p >> hwoShift;
        const int rm = p & (HWo - 1);
        const int y  = rm >> woShift, xo = rm & (Wout - 1);
        #pragma unroll
        for (int r = 0; r < 8; ++r) {
            int co = co0 + r + half * 8;
            float f = (float)acc[nt][r] * bnsc[co] + bnsh[co];
            int oi = (n * Cout + co) * HWo + y * Wout + xo;
            if (idn)  f += idn[oi];
            if (outf) outf[oi] = f;
            if (outs) outs[oi] = (f >= 0.0f) ? (signed char)1 : (signed char)-1;
        }
    }
}

__global__ __launch_bounds__(128) void brn_avgpool_fc(
    const signed char* __restrict__ s4,
    const float* __restrict__ fw, const float* __restrict__ fb,
    const float* __restrict__ scl, float* __restrict__ out) {
    __shared__ float pooled[512];
    const int n = blockIdx.x, tid = threadIdx.x;
    for (int c = tid; c < 512; c += 128) {
        const signed char* p = s4 + ((long)n * 512 + c) * 16;
        int sum = 0;
        #pragma unroll
        for (int i = 0; i < 16; ++i) sum += p[i];
        pooled[c] = (float)sum * (1.0f / 16.0f);
    }
    __syncthreads();
    if (tid < 10) {
        float d = 0.0f;
        for (int c = 0; c < 512; ++c) d += pooled[c] * fw[tid * 512 + c];
        out[(long)n * 10 + tid] = (d + fb[tid]) * scl[tid];
    }
}

extern "C" void kernel_launch(void* const* d_in, const int* in_sizes, int n_in,
                              void* d_out, int out_size, void* d_ws, size_t ws_size,
                              hipStream_t stream) {
    (void)in_sizes; (void)n_in; (void)out_size; (void)ws_size;

    char* ws = (char*)d_ws;
    size_t off = 0;
    auto alloc = [&](size_t bytes) -> char* {
        off = (off + 255) & ~(size_t)255;
        char* p = ws + off;
        off += bytes;
        return p;
    };
    auto ilog2 = [](int x) { int s = 0; while ((1 << s) < x) ++s; return s; };

    const size_t BIG = 33554432;
    float* ACT0 = (float*)alloc(BIG * 4);
    float* ACT1 = (float*)alloc(BIG * 4);
    signed char* S0 = (signed char*)alloc(BIG);
    signed char* S1 = (signed char*)alloc(BIG);
    signed char* HS = (signed char*)alloc(BIG);
    float* IDN = (float*)alloc((size_t)512 * 128 * 16 * 16 * 4);
    signed char* WPB = (signed char*)alloc((size_t)12 << 20);
    float* BNB = (float*)alloc((size_t)64 << 10);

    size_t wpoff = 0, bnoff = 0;
    int ii = 0;
    auto in = [&](void) -> const float* { return (const float*)d_in[ii++]; };

    auto foldbn = [&](const float* g, const float* b, const float* m, const float* v,
                      int C, float** sc, float** sh) {
        *sc = BNB + bnoff; bnoff += (size_t)C;
        *sh = BNB + bnoff; bnoff += (size_t)C;
        brn_fold_bn<<<dim3((C + 63) / 64), dim3(64), 0, stream>>>(g, b, m, v, C, *sc, *sh);
    };
    auto packw = [&](const float* w, int Cout, int Cin, int ks2) -> signed char* {
        signed char* o = WPB + wpoff;
        wpoff += (size_t)Cout * Cin * ks2;
        long total = (long)Cout * Cin * ks2;
        brn_pack_w<<<dim3((unsigned)((total + 255) / 256)), dim3(256), 0, stream>>>(
            w, o, Cout, Cin, ks2, ilog2(Cout), ilog2(Cin >> 6));
        return o;
    };
    auto conv = [&](const signed char* a, const signed char* wp,
                    const float* sc, const float* sh, const float* idn,
                    float* of, signed char* os,
                    int cin, int hi, int cout, int ho, int stride, int ks, int pad) {
        dim3 grid((unsigned)((512L * ho * ho) / 32), (unsigned)(cout / 64));
        const int woS = ilog2(ho), hwoS = 2 * woS;
        const int nch = cin >> 6;
#define BRN_BC(N, K) brn_binconv_wmma<N, K><<<grid, dim3(128), 0, stream>>>( \
            a, wp, sc, sh, idn, of, os, cin, hi, hi, cout, hwoS, woS, stride, pad)
        if (ks == 3) {
            if (nch == 1) BRN_BC(1, 3);
            else if (nch == 2) BRN_BC(2, 3);
            else if (nch == 4) BRN_BC(4, 3);
            else BRN_BC(8, 3);
        } else {
            if (nch == 1) BRN_BC(1, 1);
            else if (nch == 2) BRN_BC(2, 1);
            else BRN_BC(4, 1);
        }
#undef BRN_BC
    };

    const float* X  = in();
    const float* SW = in();
    const float* sg = in(); const float* sb = in();
    const float* sm = in(); const float* sv = in();
    float *ssc, *ssh;
    foldbn(sg, sb, sm, sv, 64, &ssc, &ssh);
    {
        long total = 512L * 64 * 32 * 32;
        brn_stem<<<dim3((unsigned)(total / 256)), dim3(256), 0, stream>>>(
            X, SW, ssc, ssh, ACT0, S0);
    }

    float* curF = ACT0; float* altF = ACT1;
    signed char* curS = S0; signed char* altS = S1;

    const int scin[4]  = { 64,  64, 128, 256 };
    const int scout[4] = { 64, 128, 256, 512 };
    const int sstr[4]  = {  1,   2,   2,   2 };
    int Hi = 32;

    for (int s = 0; s < 4; ++s) {
        for (int b = 0; b < 2; ++b) {
            const int cin    = (b == 0) ? scin[s] : scout[s];
            const int cout   = scout[s];
            const int stride = (b == 0) ? sstr[s] : 1;
            const int Ho     = Hi / stride;
            const bool down  = (b == 0) && (s > 0);

            const float* w1 = in();
            const float *g1 = in(), *b1 = in(), *m1 = in(), *v1 = in();
            const float* w2 = in();
            const float *g2 = in(), *b2 = in(), *m2 = in(), *v2 = in();
            const float *wd = nullptr, *gd = nullptr, *bd = nullptr,
                        *md = nullptr, *vd = nullptr;
            if (down) { wd = in(); gd = in(); bd = in(); md = in(); vd = in(); }

            signed char* wp1 = packw(w1, cout, cin, 9);
            signed char* wp2 = packw(w2, cout, cout, 9);
            signed char* wpd = down ? packw(wd, cout, cin, 1) : nullptr;
            float *sc1, *sh1, *sc2, *sh2, *scd = nullptr, *shd = nullptr;
            foldbn(g1, b1, m1, v1, cout, &sc1, &sh1);
            foldbn(g2, b2, m2, v2, cout, &sc2, &sh2);
            if (down) foldbn(gd, bd, md, vd, cout, &scd, &shd);

            conv(curS, wp1, sc1, sh1, nullptr, nullptr, HS,
                 cin, Hi, cout, Ho, stride, 3, 1);

            const float* idn = curF;
            if (down) {
                conv(curS, wpd, scd, shd, nullptr, IDN, nullptr,
                     cin, Hi, cout, Ho, stride, 1, 0);
                idn = IDN;
            }

            conv(HS, wp2, sc2, sh2, idn, altF, altS,
                 cout, Ho, cout, Ho, 1, 3, 1);

            { float* tf = curF; curF = altF; altF = tf; }
            { signed char* ts = curS; curS = altS; altS = ts; }
            Hi = Ho;
        }
    }

    const float* fcw = in();
    const float* fcb = in();
    const float* scl = in();
    brn_avgpool_fc<<<dim3(512), dim3(128), 0, stream>>>(
        curS, fcw, fcb, scl, (float*)d_out);
}
